// EventSequenceEmbedder_66632122630827
// MI455X (gfx1250) — matrix operand order repaired and checked by reading, not compiled, not run on hardware
//
#include <hip/hip_runtime.h>
#include <hip/hip_fp16.h>

typedef __attribute__((ext_vector_type(16))) _Float16 v16h;
typedef __attribute__((ext_vector_type(8)))  _Float16 v8h;
typedef __attribute__((ext_vector_type(4)))  _Float16 v4h;
typedef __attribute__((ext_vector_type(8)))  float    v8f;
typedef __attribute__((ext_vector_type(4)))  float    v4f;

namespace {
constexpr int kB       = 32;
constexpr int kS       = 1024;
constexpr int kD       = 256;
constexpr int kMP      = 9;
constexpr int kNA      = 8;
constexpr int kK       = 8 * kD;      // 2048 contraction length
constexpr int kRows    = 64;          // M rows of `combined` staged per workgroup
constexpr int kThreads = 512;         // 16 wave32 waves
// K-blocked LDS layout: 32 blocks x (64 rows x (64 K-halves + 8 pad halves)).
// Row stride 144 B: 16B-aligned for ds_load_b128, 36-word shift -> conflict-free
// fragment loads (each bank serves exactly 2 words per wave b128 load).
constexpr int kBlkK    = 64;          // K elements per LDS block
constexpr int kRowH    = kBlkK + 8;   // halves per row in a block (144 B)
constexpr int kBlkRows = kRows * kRowH;            // halves per block
constexpr int kNumBlk  = kK / kBlkK;               // 32
constexpr size_t kLdsBytes  = (size_t)kNumBlk * kBlkRows * sizeof(_Float16);  // 294912 B
constexpr size_t kWBytesF16 = (size_t)kD * kK * sizeof(_Float16);             // 1 MB
}

// One-time f32 -> f16 conversion of combine_W into workspace.
__global__ __launch_bounds__(256)
void convert_weights_f16(const float* __restrict__ w, _Float16* __restrict__ wh) {
  const int i = (blockIdx.x * 256 + threadIdx.x) * 4;
  if (i < kD * kK) {
    v4f f = *(const v4f*)(w + i);
    v4h h;
    #pragma unroll
    for (int j = 0; j < 4; ++j) h[j] = (_Float16)f[j];
    *(v4h*)(wh + i) = h;
  }
}

template <bool USE_F16W>
__global__ __launch_bounds__(kThreads)
void poker_embed_wmma(
    const int*   __restrict__ cards,       // [B,S,7]
    const int*   __restrict__ hero_pos,    // [B,S]
    const int*   __restrict__ acting_pos,  // [B,S]
    const int*   __restrict__ num_players, // [B,S]
    const float* __restrict__ scalars,     // [B,S,2]
    const float* __restrict__ blinds,      // [B,S,2]
    const float* __restrict__ bets,        // [B,S,9]
    const float* __restrict__ action,      // [B,S,8]
    const float* __restrict__ maskp,       // [B,S]
    const float* __restrict__ card_table,  // [53,D]
    const float* __restrict__ hero_table,  // [9,D]
    const float* __restrict__ acting_table,// [9,D]
    const float* __restrict__ nump_table,  // [10,D]
    const float* __restrict__ scalar_W,    // [D,2]
    const float* __restrict__ scalar_b,    // [D]
    const float* __restrict__ blind_W,     // [D,2]
    const float* __restrict__ blind_b,     // [D]
    const float* __restrict__ bet_W,       // [D,9]
    const float* __restrict__ bet_b,       // [D]
    const float* __restrict__ action_W,    // [D,8]
    const float* __restrict__ action_b,    // [D]
    const float* __restrict__ combine_W,   // [D, 8D] f32
    const _Float16* __restrict__ combine_Wh, // [D, 8D] f16 (pre-converted) or nullptr
    const float* __restrict__ combine_b,   // [D]
    float*       __restrict__ out)         // [B,S,D]
{
  extern __shared__ _Float16 lds[];
  const int tid   = threadIdx.x;
  const int mBase = blockIdx.x * kRows;

  // combined(r, k) lives at lds[(k/64)*kBlkRows + r*kRowH + (k%64)]
  auto stComb = [&](int r, int k, float v) {
    lds[(k >> 6) * kBlkRows + r * kRowH + (k & (kBlkK - 1))] = (_Float16)v;
  };

  // ---------------- Phase 1: build combined[64, 2048] tile in LDS as f16 -------------
  {
    const int seg = tid >> 6;              // concat order: card,hero,acting,scalar,bet,action,nump,blind
    const int r   = tid & 63;
    const int m   = mBase + r;
    const int k0  = seg * kD;

    switch (seg) {
      case 0: {                            // mean of 7 card embeddings
        const float* p[7];
        #pragma unroll
        for (int c = 0; c < 7; ++c) p[c] = card_table + cards[m * 7 + c] * kD;
        for (int d = 0; d < kD; ++d) {
          float s = 0.f;
          #pragma unroll
          for (int c = 0; c < 7; ++c) s += p[c][d];
          stComb(r, k0 + d, s * (1.0f / 7.0f));
        }
      } break;
      case 1: {
        const float* p = hero_table + hero_pos[m] * kD;
        for (int d = 0; d < kD; ++d) stComb(r, k0 + d, p[d]);
      } break;
      case 2: {
        const float* p = acting_table + acting_pos[m] * kD;
        for (int d = 0; d < kD; ++d) stComb(r, k0 + d, p[d]);
      } break;
      case 3: {                            // scalar linear, K=2
        const float s0 = scalars[2 * m], s1 = scalars[2 * m + 1];
        for (int d = 0; d < kD; ++d)
          stComb(r, k0 + d, s0 * scalar_W[2 * d] + s1 * scalar_W[2 * d + 1] + scalar_b[d]);
      } break;
      case 4: {                            // bet linear, K=9
        float bv[kMP];
        #pragma unroll
        for (int j = 0; j < kMP; ++j) bv[j] = bets[m * kMP + j];
        for (int d = 0; d < kD; ++d) {
          float s = bet_b[d];
          #pragma unroll
          for (int j = 0; j < kMP; ++j) s += bv[j] * bet_W[d * kMP + j];
          stComb(r, k0 + d, s);
        }
      } break;
      case 5: {                            // action linear, K=8
        float av[kNA];
        #pragma unroll
        for (int j = 0; j < kNA; ++j) av[j] = action[m * kNA + j];
        for (int d = 0; d < kD; ++d) {
          float s = action_b[d];
          #pragma unroll
          for (int j = 0; j < kNA; ++j) s += av[j] * action_W[d * kNA + j];
          stComb(r, k0 + d, s);
        }
      } break;
      case 6: {
        const float* p = nump_table + num_players[m] * kD;
        for (int d = 0; d < kD; ++d) stComb(r, k0 + d, p[d]);
      } break;
      default: {                           // blind linear, K=2
        const float b0 = blinds[2 * m], b1 = blinds[2 * m + 1];
        for (int d = 0; d < kD; ++d)
          stComb(r, k0 + d, b0 * blind_W[2 * d] + b1 * blind_W[2 * d + 1] + blind_b[d]);
      } break;
    }
  }
  __syncthreads();

  // ---------------- Phase 2: out = combined @ combine_W^T via v_wmma_f32_16x16x32_f16 ----
  const int wave  = tid >> 5;              // 0..15 -> owns N-tile [wave*16, wave*16+16)
  const int lane  = tid & 31;
  const int l16   = lane & 15;
  const int hi    = lane >> 4;             // lane half selects K sub-range per ISA layout
  const int n     = wave * 16 + l16;       // output column of this lane's B fragment

  const float*    wrow  = combine_W + (size_t)n * kK;
  const _Float16* wrowh = USE_F16W ? (combine_Wh + (size_t)n * kK) : nullptr;

  // B fragment (32x16 f16): lanes 0-15 hold K {kb..+7, kb+16..+23}, lanes 16-31 hold {kb+8..+15, kb+24..+31}
  auto loadB = [&](int kb) -> v16h {
    if constexpr (USE_F16W) {
      const v8h* s = (const v8h*)(wrowh + kb + 8 * hi);     // 16B aligned
      return __builtin_shufflevector(s[0], s[2],
                                     0, 1, 2, 3, 4, 5, 6, 7,
                                     8, 9, 10, 11, 12, 13, 14, 15);
    } else {
      const v4f* s = (const v4f*)(wrow + kb + 8 * hi);
      v16h b;
      #pragma unroll
      for (int q = 0; q < 4; ++q) {
        v4f f = s[q + (q & 2)];
        #pragma unroll
        for (int j = 0; j < 4; ++j) b[4 * q + j] = (_Float16)f[j];
      }
      return b;
    }
  };
  // A fragment; 16-bit A layout: lane<16 -> K {kb..+7, kb+16..+23},
  // lane>=16 -> K {kb+8..+15, kb+24..+31}; row M = lane&15.
  auto loadA = [&](int kb, int t) -> v16h {
    const _Float16* p = &lds[(kb >> 6) * kBlkRows + (t * 16 + l16) * kRowH +
                             (kb & (kBlkK - 1)) + 8 * hi];
    v8h lo = *(const v8h*)(p);              // 16B-aligned ds_load_b128
    v8h hh = *(const v8h*)(p + 16);
    return __builtin_shufflevector(lo, hh,
                                   0, 1, 2, 3, 4, 5, 6, 7,
                                   8, 9, 10, 11, 12, 13, 14, 15);
  };

  v8f acc[4] = {};                          // 4 M-tiles x 16x16 f32 accumulators
  v16h a0[4], a1[4], b0, b1;

  b0 = loadB(0);
  #pragma unroll
  for (int t = 0; t < 4; ++t) a0[t] = loadA(0, t);

  for (int kb = 0; kb < kK; kb += 64) {
    // Stage 1 loads (disjoint regs -> no WAR hazard with stage-0 WMMAs)
    b1 = loadB(kb + 32);
    #pragma unroll
    for (int t = 0; t < 4; ++t) a1[t] = loadA(kb + 32, t);
    if (kb + 128 < kK)
      __builtin_prefetch(USE_F16W ? (const void*)(wrowh + kb + 128 + 16 * hi)
                                  : (const void*)(wrow + kb + 128 + 16 * hi), 0, 1);

    #pragma unroll
    for (int t = 0; t < 4; ++t)
      acc[t] = __builtin_amdgcn_wmma_f32_16x16x32_f16(false, a0[t], false, b0,
                                                      (short)0, acc[t], false, false);
    // Refill stage-0 buffers for the next 64-K block (skip on last iteration)
    if (kb + 64 < kK) {
      b0 = loadB(kb + 64);
      #pragma unroll
      for (int t = 0; t < 4; ++t) a0[t] = loadA(kb + 64, t);
    }
    #pragma unroll
    for (int t = 0; t < 4; ++t)
      acc[t] = __builtin_amdgcn_wmma_f32_16x16x32_f16(false, a1[t], false, b1,
                                                      (short)0, acc[t], false, false);
  }

  // ---------------- Epilogue: bias + mask; C/D layout: VGPR v -> M = v + 8*hi, lane%16 -> N
  const float bias = combine_b[n];
  #pragma unroll
  for (int t = 0; t < 4; ++t) {
    #pragma unroll
    for (int v = 0; v < 8; ++v) {
      const int m = mBase + t * 16 + hi * 8 + v;
      out[(size_t)m * kD + n] = (acc[t][v] + bias) * maskp[m];
    }
  }
}

extern "C" void kernel_launch(void* const* d_in, const int* in_sizes, int n_in,
                              void* d_out, int out_size, void* d_ws, size_t ws_size,
                              hipStream_t stream) {
  (void)in_sizes; (void)n_in; (void)out_size;
  const dim3 grid((kB * kS) / kRows);       // 512 workgroups

  const bool useF16W = (d_ws != nullptr) && (ws_size >= kWBytesF16);

  if (useF16W) {
    convert_weights_f16<<<dim3((kD * kK) / (256 * 4)), 256, 0, stream>>>(
        (const float*)d_in[21], (_Float16*)d_ws);
    poker_embed_wmma<true><<<grid, kThreads, kLdsBytes, stream>>>(
        (const int*)d_in[0],  (const int*)d_in[1],  (const int*)d_in[2],  (const int*)d_in[3],
        (const float*)d_in[4], (const float*)d_in[5], (const float*)d_in[6], (const float*)d_in[7],
        (const float*)d_in[8],
        (const float*)d_in[9], (const float*)d_in[10], (const float*)d_in[11], (const float*)d_in[12],
        (const float*)d_in[13], (const float*)d_in[14], (const float*)d_in[15], (const float*)d_in[16],
        (const float*)d_in[17], (const float*)d_in[18], (const float*)d_in[19], (const float*)d_in[20],
        (const float*)d_in[21], (const _Float16*)d_ws, (const float*)d_in[22],
        (float*)d_out);
  } else {
    poker_embed_wmma<false><<<grid, kThreads, kLdsBytes, stream>>>(
        (const int*)d_in[0],  (const int*)d_in[1],  (const int*)d_in[2],  (const int*)d_in[3],
        (const float*)d_in[4], (const float*)d_in[5], (const float*)d_in[6], (const float*)d_in[7],
        (const float*)d_in[8],
        (const float*)d_in[9], (const float*)d_in[10], (const float*)d_in[11], (const float*)d_in[12],
        (const float*)d_in[13], (const float*)d_in[14], (const float*)d_in[15], (const float*)d_in[16],
        (const float*)d_in[17], (const float*)d_in[18], (const float*)d_in[19], (const float*)d_in[20],
        (const float*)d_in[21], nullptr, (const float*)d_in[22],
        (float*)d_out);
  }
}
